// MambaEmbed_31533649887455
// MI455X (gfx1250) — matrix operand order repaired and checked by reading, not compiled, not run on hardware
//
#include <hip/hip_runtime.h>
#include <math.h>

typedef __bf16 bf16_t;
typedef __attribute__((ext_vector_type(16))) __bf16 v16bf;
typedef __attribute__((ext_vector_type(8)))  __bf16 v8bf;
typedef __attribute__((ext_vector_type(8)))  float  v8f;

namespace {
constexpr int kBB   = 16;
constexpr int kT    = 2048;
constexpr int kDin  = 64;
constexpr int kD    = 512;
constexpr int kDI   = 1024;   // 2*kD
constexpr int kN    = 16;     // SSM state size
constexpr int kDC   = 4;      // conv width
constexpr int kR    = 32;     // dt rank
constexpr int kL    = 3;
constexpr int kDout = 128;
constexpr int kM    = kBB * kT;  // 32768 rows in every GEMM

// GEMM tiling: block tile 128x64 (8 waves, 4x2; each wave 2x2 WMMA tiles),
// K-stage 64 (= two 16x16x32 WMMA K-steps per barrier), double-buffered LDS
// filled by GLOBAL_LOAD_ASYNC_TO_LDS_B128.
constexpr int BM   = 128;
constexpr int BN   = 64;
constexpr int BK   = 64;
constexpr int LDSS = 72;   // LDS row stride in bf16 (144B): 16B-aligned rows,
                           // 36-bank stride -> conflict-free 16-row b128 reads
}

// -------------------------------------------------------------------------
// CDNA5 async copy: 16 bytes global -> LDS per lane, tracked by ASYNCcnt.
// -------------------------------------------------------------------------
__device__ __forceinline__ void async_copy_b128(unsigned lds_addr, const void* gaddr) {
  asm volatile("global_load_async_to_lds_b128 %0, %1, off"
               :: "v"(lds_addr), "v"((unsigned long long)(size_t)gaddr)
               : "memory");
}

__device__ __forceinline__ void wait_async0() {
#if __has_builtin(__builtin_amdgcn_s_wait_asynccnt)
  __builtin_amdgcn_s_wait_asynccnt(0);
#else
  asm volatile("s_wait_asynccnt 0x0" ::: "memory");
#endif
}

// -------------------------------------------------------------------------
// Generic C[M,N] = act( A[M,K](lda) @ W[N,K]^T + bias ), A/W bf16 in HBM,
// f32 accumulate via v_wmma_f32_16x16x32_bf16. TOUT = bf16 or float.
// ACT: 0 = none, 1 = +bias, 2 = softplus(x+bias), 3 = tanh(x+bias)
// -------------------------------------------------------------------------
template<int ACT, typename TOUT>
__global__ __launch_bounds__(256) void gemm_bf16_wmma(
    const bf16_t* __restrict__ A, const bf16_t* __restrict__ W,
    const float* __restrict__ bias, TOUT* __restrict__ C,
    int M, int N, int K, int lda, int ldc)
{
  __shared__ alignas(16) bf16_t As[2][BM * LDSS];
  __shared__ alignas(16) bf16_t Ws[2][BN * LDSS];

  const int tid  = threadIdx.x;
  const int lane = tid & 31;
  const int wave = tid >> 5;
  const int wm   = wave >> 1;          // 0..3 -> 32-row slab
  const int wn   = wave & 1;           // 0..1 -> 32-col slab
  const int block_m = blockIdx.y * BM;
  const int block_n = blockIdx.x * BN;

  const int half = lane >> 4;          // 0: lanes 0-15, 1: lanes 16-31
  const int l16  = lane & 15;

  v8f acc[2][2];
#pragma unroll
  for (int mi = 0; mi < 2; ++mi)
#pragma unroll
    for (int ni = 0; ni < 2; ++ni)
#pragma unroll
      for (int v = 0; v < 8; ++v) acc[mi][ni][v] = 0.0f;

  // Async stage of one K-slab (BMxBK of A, BNxBK of W) into LDS buffer `buf`.
  auto stage = [&](int k0, int buf) {
    // A tile: 128 rows x 64 cols -> 1024 b128 chunks, 4 per thread.
#pragma unroll
    for (int it = 0; it < 4; ++it) {
      const int c    = tid + it * 256;
      const int r    = c >> 3;
      const int col8 = (c & 7) << 3;   // element col of 8-elem chunk
      const bf16_t* g = A + (size_t)(block_m + r) * lda + (k0 + col8);
      async_copy_b128((unsigned)(size_t)&As[buf][r * LDSS + col8], g);
    }
    // W tile: 64 rows x 64 cols -> 512 chunks, 2 per thread.
#pragma unroll
    for (int it = 0; it < 2; ++it) {
      const int c    = tid + it * 256;
      const int r    = c >> 3;
      const int col8 = (c & 7) << 3;
      const bf16_t* g = W + (size_t)(block_n + r) * K + (k0 + col8);
      async_copy_b128((unsigned)(size_t)&Ws[buf][r * LDSS + col8], g);
    }
  };

  stage(0, 0);
  int buf = 0;
  for (int k0 = 0; k0 < K; k0 += BK) {
    wait_async0();
    __syncthreads();
    if (k0 + BK < K) stage(k0 + BK, buf ^ 1);

#pragma unroll
    for (int ks = 0; ks < BK; ks += 32) {
      if (k0 + ks < K) {
        // A frags: lane<16 -> K {ks+0..7, ks+16..23}; lane>=16 -> {+8..15, +24..31}
        v16bf a_frag[2];
#pragma unroll
        for (int mi = 0; mi < 2; ++mi) {
          const bf16_t* ap = &As[buf][(wm * 32 + mi * 16 + l16) * LDSS + ks + (half ? 8 : 0)];
          union { v16bf v; v8bf h[2]; } u;
          u.h[0] = *(const v8bf*)(ap);
          u.h[1] = *(const v8bf*)(ap + 16);
          a_frag[mi] = u.v;
        }
        // B frags: lane<16 -> K {ks+0..7, ks+16..23}; lane>=16 -> {+8..15, +24..31}
        v16bf b_frag[2];
#pragma unroll
        for (int ni = 0; ni < 2; ++ni) {
          const bf16_t* bp = &Ws[buf][(wn * 32 + ni * 16 + l16) * LDSS + ks + (half ? 8 : 0)];
          union { v16bf v; v8bf h[2]; } u;
          u.h[0] = *(const v8bf*)(bp);
          u.h[1] = *(const v8bf*)(bp + 16);
          b_frag[ni] = u.v;
        }
#pragma unroll
        for (int mi = 0; mi < 2; ++mi)
#pragma unroll
          for (int ni = 0; ni < 2; ++ni)
            acc[mi][ni] = __builtin_amdgcn_wmma_f32_16x16x32_bf16(
                false, a_frag[mi], false, b_frag[ni],
                (short)0, acc[mi][ni], false, false);
      }
    }
    buf ^= 1;
  }

  // Epilogue. D layout: VGPR v, lane l -> row = v + 8*(l>=16), col = l&15.
#pragma unroll
  for (int mi = 0; mi < 2; ++mi) {
#pragma unroll
    for (int ni = 0; ni < 2; ++ni) {
      const int col = block_n + wn * 32 + ni * 16 + l16;
      float bv = 0.0f;
      if (ACT != 0) bv = bias[col];
#pragma unroll
      for (int v = 0; v < 8; ++v) {
        const int row = block_m + wm * 32 + mi * 16 + half * 8 + v;
        float val = acc[mi][ni][v] + bv;
        if (ACT == 2) val = (val > 20.0f) ? val : log1pf(__expf(val));
        if (ACT == 3) val = tanhf(val);
        C[(size_t)row * ldc + col] = (TOUT)val;
      }
    }
  }
}

// -------------------------------------------------------------------------
// One-time f32 -> bf16 conversion (weights and x).
// -------------------------------------------------------------------------
__global__ __launch_bounds__(256) void cvt_f32_bf16(
    const float* __restrict__ s, bf16_t* __restrict__ d, int n)
{
  const int i = blockIdx.x * 256 + threadIdx.x;
  if (i < n) d[i] = (bf16_t)s[i];
}

// -------------------------------------------------------------------------
// Depthwise causal conv (width 4) + bias + SiLU over xin = xz[:, :, :DI].
// -------------------------------------------------------------------------
__global__ __launch_bounds__(256) void conv_silu_kernel(
    const bf16_t* __restrict__ xz, const float* __restrict__ cw,
    const float* __restrict__ cb, bf16_t* __restrict__ u)
{
  const int idx = blockIdx.x * 256 + threadIdx.x;   // [0, kM*kDI)
  const int d  = idx & (kDI - 1);
  const int bt = idx >> 10;                         // b*kT + t
  const int t  = bt & (kT - 1);

  float acc = cb[d];
#pragma unroll
  for (int k = 0; k < kDC; ++k) {
    const int tt = t - (kDC - 1) + k;
    if (tt >= 0)
      acc += (float)xz[(size_t)(bt - (kDC - 1) + k) * (2 * kDI) + d] * cw[d * kDC + k];
  }
  u[idx] = (bf16_t)(acc / (1.0f + __expf(-acc)));   // silu
}

// -------------------------------------------------------------------------
// Selective scan + skip + gate, fused. One thread per (b,d); 16-wide f32
// state in registers; B_t/C_t broadcast via double-buffered LDS (one barrier
// per step). Writes y * silu(z) in place over u.
// -------------------------------------------------------------------------
__global__ __launch_bounds__(256) void scan_gate_kernel(
    const bf16_t* __restrict__ delta, const bf16_t* __restrict__ dbc,
    const bf16_t* __restrict__ xz,    const float*  __restrict__ A_log,
    const float*  __restrict__ Dp,    bf16_t* __restrict__ u)
{
  __shared__ float sB[2][kN];
  __shared__ float sC[2][kN];

  const int tid = threadIdx.x;
  const int b   = blockIdx.x >> 2;                  // kDI/256 == 4 blocks/batch
  const int d   = ((blockIdx.x & 3) << 8) + tid;

  float Areg[kN];
#pragma unroll
  for (int n = 0; n < kN; ++n) Areg[n] = -__expf(A_log[d * kN + n]);
  const float Dv = Dp[d];

  float h[kN];
#pragma unroll
  for (int n = 0; n < kN; ++n) h[n] = 0.0f;

  for (int t = 0; t < kT; ++t) {
    const int buf = t & 1;
    const size_t row = (size_t)(b * kT + t);
    if (tid < kN)           sB[buf][tid]      = (float)dbc[row * 64 + kR + tid];
    else if (tid < 2 * kN)  sC[buf][tid - kN] = (float)dbc[row * 64 + kR + kN + (tid - kN)];
    __syncthreads();

    const float dv = (float)delta[row * kDI + d];
    const float uv = (float)u[row * kDI + d];
    const float du = dv * uv;
    float y = 0.0f;
#pragma unroll
    for (int n = 0; n < kN; ++n) {
      const float dA = __expf(dv * Areg[n]);
      h[n] = dA * h[n] + du * sB[buf][n];
      y += h[n] * sC[buf][n];
    }
    y += uv * Dv;
    const float z  = (float)xz[row * (2 * kDI) + kDI + d];
    const float gz = z / (1.0f + __expf(-z));       // silu(z)
    u[row * kDI + d] = (bf16_t)(y * gz);
  }
}

// -------------------------------------------------------------------------
extern "C" void kernel_launch(void* const* d_in, const int* in_sizes, int n_in,
                              void* d_out, int out_size, void* d_ws, size_t ws_size,
                              hipStream_t stream)
{
  const float* x     = (const float*)d_in[0];
  const float* Wi    = (const float*)d_in[1];
  const float* bi    = (const float*)d_in[2];
  const float* Win   = (const float*)d_in[3];
  const float* convw = (const float*)d_in[4];
  const float* convb = (const float*)d_in[5];
  const float* Wx    = (const float*)d_in[6];
  const float* Wdt   = (const float*)d_in[7];
  const float* bdt   = (const float*)d_in[8];
  const float* A_log = (const float*)d_in[9];
  const float* Dp    = (const float*)d_in[10];
  const float* Wout  = (const float*)d_in[11];
  const float* Wo    = (const float*)d_in[12];
  const float* bo    = (const float*)d_in[13];
  float* out = (float*)d_out;

  // Workspace: everything bf16 (~320 MB). All chunk sizes are multiples of 8
  // elements, keeping every buffer 16B-aligned for b128 async copies.
  bf16_t* bws = (bf16_t*)d_ws;
  size_t off = 0;
  auto alloc = [&](size_t n) { bf16_t* p = bws + off; off += n; return p; };
  bf16_t* xb    = alloc((size_t)kM * kDin);
  bf16_t* Wib   = alloc((size_t)kD * kDin);
  bf16_t* Winb  = alloc((size_t)kL * 2 * kDI * kD);
  bf16_t* Wxb   = alloc((size_t)kL * 64 * kDI);
  bf16_t* Wdtb  = alloc((size_t)kL * kDI * kR);
  bf16_t* Woutb = alloc((size_t)kL * kD * kDI);
  bf16_t* Wob   = alloc((size_t)kDout * kD);
  bf16_t* h     = alloc((size_t)kM * kD);
  bf16_t* xz    = alloc((size_t)kM * 2 * kDI);
  bf16_t* u     = alloc((size_t)kM * kDI);
  bf16_t* dbc   = alloc((size_t)kM * 64);
  bf16_t* delta = alloc((size_t)kM * kDI);

  const dim3 blk(256);
  auto cvt = [&](const float* s, bf16_t* d, size_t n) {
    cvt_f32_bf16<<<(unsigned)((n + 255) / 256), blk, 0, stream>>>(s, d, (int)n);
  };
  cvt(x,    xb,    (size_t)kM * kDin);
  cvt(Wi,   Wib,   (size_t)kD * kDin);
  cvt(Win,  Winb,  (size_t)kL * 2 * kDI * kD);
  cvt(Wx,   Wxb,   (size_t)kL * 64 * kDI);
  cvt(Wdt,  Wdtb,  (size_t)kL * kDI * kR);
  cvt(Wout, Woutb, (size_t)kL * kD * kDI);
  cvt(Wo,   Wob,   (size_t)kDout * kD);

  const int gm = kM / BM;   // 256

  // h = x @ Wi^T + bi   (N=512, K=64)
  gemm_bf16_wmma<1, bf16_t><<<dim3(kD / BN, gm), blk, 0, stream>>>(
      xb, Wib, bi, h, kM, kD, kDin, kDin, kD);

  for (int l = 0; l < kL; ++l) {
    const bf16_t* Win_l  = Winb  + (size_t)l * (2 * kDI) * kD;
    const float*  cw_l   = convw + (size_t)l * kDI * kDC;
    const float*  cb_l   = convb + (size_t)l * kDI;
    const bf16_t* Wx_l   = Wxb   + (size_t)l * 64 * kDI;
    const bf16_t* Wdt_l  = Wdtb  + (size_t)l * kDI * kR;
    const float*  bdt_l  = bdt   + (size_t)l * kDI;
    const float*  Alog_l = A_log + (size_t)l * kDI * kN;
    const float*  Dp_l   = Dp    + (size_t)l * kDI;
    const bf16_t* Wout_l = Woutb + (size_t)l * kD * kDI;

    // xz = h @ Win^T   (N=2048, K=512)
    gemm_bf16_wmma<0, bf16_t><<<dim3(2 * kDI / BN, gm), blk, 0, stream>>>(
        h, Win_l, nullptr, xz, kM, 2 * kDI, kD, kD, 2 * kDI);

    // u = silu(conv(xin) + cb)
    conv_silu_kernel<<<(kM * kDI) / 256, blk, 0, stream>>>(xz, cw_l, cb_l, u);

    // dbc = u @ Wx^T   (N=64, K=1024)
    gemm_bf16_wmma<0, bf16_t><<<dim3(64 / BN, gm), blk, 0, stream>>>(
        u, Wx_l, nullptr, dbc, kM, 64, kDI, kDI, 64);

    // delta = softplus(dt @ Wdt^T + bdt); dt = dbc[:, :32] (lda=64, K=32)
    gemm_bf16_wmma<2, bf16_t><<<dim3(kDI / BN, gm), blk, 0, stream>>>(
        dbc, Wdt_l, bdt_l, delta, kM, kDI, kR, 64, kDI);

    // selective scan + skip + gate; (y * silu(z)) written over u
    scan_gate_kernel<<<kBB * (kDI / 256), blk, 0, stream>>>(
        delta, dbc, xz, Alog_l, Dp_l, u);

    // h = yg @ Wout^T   (N=512, K=1024)
    gemm_bf16_wmma<0, bf16_t><<<dim3(kD / BN, gm), blk, 0, stream>>>(
        u, Wout_l, nullptr, h, kM, kD, kDI, kDI, kD);
  }

  // out = tanh(h @ Wo^T + bo)   (N=128, K=512), f32 output
  gemm_bf16_wmma<3, float><<<dim3(kDout / BN, gm), blk, 0, stream>>>(
      h, Wob, bo, out, kM, kDout, kD, kD, kDout);
}
